// GCN5_39573828665577
// MI455X (gfx1250) — compile-verified, matrix operand order checked by reading, not run on hardware
//
#include <hip/hip_runtime.h>
#include <hip/hip_bf16.h>

// ---------------------------------------------------------------------------
// GCN 5-layer forward on gfx1250 (MI455X). GEMMs on v_wmma_f32_16x16x32_bf16
// with b128 fragment loads (weights pre-transposed so both A and B fragments
// are contiguous 16-byte runs). Edge scatter via global_atomic_add_f32.
// ---------------------------------------------------------------------------

#define N_NODES   50000
#define N_EDGES   500000
#define FEAT      128
#define N_CLASSES 32
#define N_GRAPHS  64

typedef __attribute__((ext_vector_type(16))) __bf16 v16bf;
typedef __attribute__((ext_vector_type(8)))  __bf16 v8bf;
typedef __attribute__((ext_vector_type(8)))  float  v8f;

// ---------------- utility kernels ----------------

__global__ void fill_f32(float* p, float v, int n) {
    int i = blockIdx.x * blockDim.x + threadIdx.x;
    if (i < n) p[i] = v;
}

__global__ void cvt_f32_to_bf16(const float* __restrict__ in, __bf16* __restrict__ out, int n) {
    int i = blockIdx.x * blockDim.x + threadIdx.x;
    if (i < n) out[i] = (__bf16)in[i];
}

// W[k][n] (f32, row-major) -> Wt[n][k] (bf16): makes B fragments contiguous.
__global__ void transpose_cvt_w(const float* __restrict__ W, __bf16* __restrict__ Wt) {
    int idx = blockIdx.x * blockDim.x + threadIdx.x;   // FEAT*FEAT threads
    if (idx >= FEAT * FEAT) return;
    int n = idx >> 7;          // output row   (col of W)
    int k = idx & 127;         // output col   (row of W)
    Wt[idx] = (__bf16)W[k * FEAT + n];
}

// deg initialized to 1.0 (self loop); add 1 per incoming edge at dst
__global__ void edge_degree(const int* __restrict__ dst, float* __restrict__ deg, int nE) {
    int e = blockIdx.x * blockDim.x + threadIdx.x;
    if (e < nE) atomicAdd(&deg[dst[e]], 1.0f);
}

__global__ void make_dis(const float* __restrict__ deg, float* __restrict__ dis, int n) {
    int i = blockIdx.x * blockDim.x + threadIdx.x;
    if (i < n) dis[i] = rsqrtf(deg[i]);
}

__global__ void make_norm_edge(const int* __restrict__ src, const int* __restrict__ dst,
                               const float* __restrict__ dis, float* __restrict__ ne, int nE) {
    int e = blockIdx.x * blockDim.x + threadIdx.x;
    if (e < nE) ne[e] = dis[src[e]] * dis[dst[e]];
}

// ---------------- WMMA GEMM: C[M,128] = A[M,128] @ Wt^T ----------------
// A    : [M, 128] bf16 row-major (activation)
// Wt   : [128, 128] bf16, Wt[n][k] = W[k][n]  (so both fragments load
//        as two contiguous 16-byte runs per lane: K base+0..7, base+16..23)
// One wave -> one 16x16 tile. 256-thread block = 8 waves = full 128-wide row
// slab of 16 nodes. K = 128 -> 4 WMMA steps.

__device__ __forceinline__ v16bf load_frag_row(const __bf16* row) {
    // row points at element K-base of a 128-element row; 16-byte aligned.
    const v8bf lo = *reinterpret_cast<const v8bf*>(row);        // K base+0..7
    const v8bf hi = *reinterpret_cast<const v8bf*>(row + 16);   // K base+16..23
    return __builtin_shufflevector(lo, hi, 0, 1, 2, 3, 4, 5, 6, 7,
                                           8, 9, 10, 11, 12, 13, 14, 15);
}

__global__ void gemm_bf16_wmma(const __bf16* __restrict__ A,
                               const __bf16* __restrict__ Wt,
                               float* __restrict__ C) {
    const int wave  = threadIdx.x >> 5;          // 0..7 -> N tile
    const int lane  = threadIdx.x & 31;
    const int half  = lane >> 4;                 // K sub-block (+8)
    const int ln15  = lane & 15;                 // A: row in tile, B/C: col
    const int mBase = blockIdx.x * 16;
    const int n     = wave * 16 + ln15;          // output column

    const __bf16* Arow = A  + (size_t)(mBase + ln15) * FEAT;
    const __bf16* Brow = Wt + (size_t)n * FEAT;

    v8f c = {};
    #pragma unroll
    for (int kc = 0; kc < 4; ++kc) {
        const int base = kc * 32 + half * 8;
        const v16bf a = load_frag_row(Arow + base);
        const v16bf b = load_frag_row(Brow + base);
        c = __builtin_amdgcn_wmma_f32_16x16x32_bf16(
                /*neg_a=*/false, a, /*neg_b=*/false, b,
                /*c_mod=*/(short)0, c, /*reuse_a=*/false, /*reuse_b=*/false);
    }

    #pragma unroll
    for (int r = 0; r < 8; ++r) {
        const int m = r + 8 * half;              // C layout: VGPR r -> M
        C[(size_t)(mBase + m) * FEAT + n] = c[r];
    }
}

// ---------------- edge scatter: agg[dst] += norm_edge * h[src] ----------------
// One edge per wave; each lane handles 4 consecutive features (float4 gather).

__global__ void scatter_edges(const float* __restrict__ h,
                              const float* __restrict__ ne,
                              const int* __restrict__ src,
                              const int* __restrict__ dst,
                              float* __restrict__ agg, int nE) {
    const int e    = (blockIdx.x * blockDim.x + threadIdx.x) >> 5;
    const int lane = threadIdx.x & 31;
    if (e >= nE) return;
    const int   s = src[e];
    const int   d = dst[e];
    const float w = ne[e];
    const float4 v = ((const float4*)(h + (size_t)s * FEAT))[lane];
    float* ap = agg + (size_t)d * FEAT + lane * 4;
    atomicAdd(ap + 0, v.x * w);
    atomicAdd(ap + 1, v.y * w);
    atomicAdd(ap + 2, v.z * w);
    atomicAdd(ap + 3, v.w * w);
}

// ---------------- combine: out = agg + h*norm_self + bias (opt ReLU) ----------
// Emits bf16 activation for the next layer's WMMA GEMM; for the last layer
// also rewrites agg in-place with the fp32 result (elementwise, safe).

__global__ void combine(const float* __restrict__ h,
                        float* __restrict__ agg,
                        const float* __restrict__ dis,
                        const float* __restrict__ bias,
                        __bf16* __restrict__ out_bf,
                        int do_relu, int write_f32) {
    const int idx = blockIdx.x * blockDim.x + threadIdx.x;
    if (idx >= N_NODES * FEAT) return;
    const int i = idx >> 7;            // node
    const int f = idx & 127;           // feature
    const float ds = dis[i];
    float v = agg[idx] + h[idx] * (ds * ds) + bias[f];
    if (do_relu) v = fmaxf(v, 0.0f);
    out_bf[idx] = (__bf16)v;
    if (write_f32) agg[idx] = v;
}

// ---------------- global mean pool (sum + count), then final linear ----------

__global__ void pool_nodes(const float* __restrict__ h,
                           const int* __restrict__ batch,
                           float* __restrict__ pool,   // [N_GRAPHS, FEAT]
                           float* __restrict__ cnt) {  // [N_GRAPHS]
    const int i    = (blockIdx.x * blockDim.x + threadIdx.x) >> 5;
    const int lane = threadIdx.x & 31;
    if (i >= N_NODES) return;
    const int g = batch[i];
    const float4 v = ((const float4*)(h + (size_t)i * FEAT))[lane];
    float* pp = pool + (size_t)g * FEAT + lane * 4;
    atomicAdd(pp + 0, v.x);
    atomicAdd(pp + 1, v.y);
    atomicAdd(pp + 2, v.z);
    atomicAdd(pp + 3, v.w);
    if (lane == 0) atomicAdd(&cnt[g], 1.0f);
}

__global__ void final_linear(const float* __restrict__ pool,
                             const float* __restrict__ cnt,
                             const float* __restrict__ Wl,   // [FEAT, N_CLASSES]
                             const float* __restrict__ bl,
                             float* __restrict__ out) {      // [N_GRAPHS, N_CLASSES]
    const int idx = blockIdx.x * blockDim.x + threadIdx.x;
    if (idx >= N_GRAPHS * N_CLASSES) return;
    const int g = idx >> 5;
    const int c = idx & 31;
    const float cinv = 1.0f / fmaxf(cnt[g], 1.0f);
    float acc = 0.0f;
    const float* pr = pool + (size_t)g * FEAT;
    #pragma unroll 8
    for (int k = 0; k < FEAT; ++k) acc = fmaf(pr[k], Wl[k * N_CLASSES + c], acc);
    out[idx] = acc * cinv + bl[c];
}

// ---------------------------------------------------------------------------

static inline size_t ws_take(size_t& off, size_t bytes) {
    size_t r = off;
    off += (bytes + 255) & ~(size_t)255;
    return r;
}

extern "C" void kernel_launch(void* const* d_in, const int* in_sizes, int n_in,
                              void* d_out, int out_size, void* d_ws, size_t ws_size,
                              hipStream_t stream) {
    (void)in_sizes; (void)n_in; (void)out_size; (void)ws_size;

    const float* x    = (const float*)d_in[0];
    const int*   eidx = (const int*)d_in[1];     // [2, N_EDGES] flat
    const int*   batch= (const int*)d_in[2];
    const float* W[5] = {(const float*)d_in[3], (const float*)d_in[5], (const float*)d_in[7],
                         (const float*)d_in[9], (const float*)d_in[11]};
    const float* b[5] = {(const float*)d_in[4], (const float*)d_in[6], (const float*)d_in[8],
                         (const float*)d_in[10], (const float*)d_in[12]};
    const float* Wlin = (const float*)d_in[13];
    const float* blin = (const float*)d_in[14];
    float* out = (float*)d_out;

    const int* src = eidx;
    const int* dst = eidx + N_EDGES;

    // --- workspace carve-out ---
    char* ws = (char*)d_ws;
    size_t off = 0;
    float*  deg   = (float*) (ws + ws_take(off, N_NODES * sizeof(float)));
    float*  dis   = (float*) (ws + ws_take(off, N_NODES * sizeof(float)));
    float*  ne    = (float*) (ws + ws_take(off, N_EDGES * sizeof(float)));
    float*  h     = (float*) (ws + ws_take(off, (size_t)N_NODES * FEAT * sizeof(float)));
    float*  agg   = (float*) (ws + ws_take(off, (size_t)N_NODES * FEAT * sizeof(float)));
    __bf16* actA  = (__bf16*)(ws + ws_take(off, (size_t)N_NODES * FEAT * sizeof(__bf16)));
    __bf16* actB  = (__bf16*)(ws + ws_take(off, (size_t)N_NODES * FEAT * sizeof(__bf16)));
    __bf16* WbT[5];
    for (int l = 0; l < 5; ++l)
        WbT[l] = (__bf16*)(ws + ws_take(off, FEAT * FEAT * sizeof(__bf16)));
    float*  pool  = (float*) (ws + ws_take(off, N_GRAPHS * FEAT * sizeof(float)));
    float*  cnt   = (float*) (ws + ws_take(off, N_GRAPHS * sizeof(float)));

    const int NT = 256;
    const int nodeElems = N_NODES * FEAT;

    // --- normalization coefficients ---
    fill_f32<<<(N_NODES + NT - 1) / NT, NT, 0, stream>>>(deg, 1.0f, N_NODES);
    edge_degree<<<(N_EDGES + NT - 1) / NT, NT, 0, stream>>>(dst, deg, N_EDGES);
    make_dis<<<(N_NODES + NT - 1) / NT, NT, 0, stream>>>(deg, dis, N_NODES);
    make_norm_edge<<<(N_EDGES + NT - 1) / NT, NT, 0, stream>>>(src, dst, dis, ne, N_EDGES);

    // --- bf16 staging: activations + transposed weights ---
    cvt_f32_to_bf16<<<(nodeElems + NT - 1) / NT, NT, 0, stream>>>(x, actA, nodeElems);
    for (int l = 0; l < 5; ++l)
        transpose_cvt_w<<<(FEAT * FEAT + NT - 1) / NT, NT, 0, stream>>>(W[l], WbT[l]);

    // --- 5 GCN layers ---
    __bf16* cur = actA;
    __bf16* nxt = actB;
    for (int l = 0; l < 5; ++l) {
        gemm_bf16_wmma<<<N_NODES / 16, 256, 0, stream>>>(cur, WbT[l], h);
        fill_f32<<<(nodeElems + NT - 1) / NT, NT, 0, stream>>>(agg, 0.0f, nodeElems);
        scatter_edges<<<(N_EDGES * 32 + NT - 1) / NT, NT, 0, stream>>>(h, ne, src, dst, agg, N_EDGES);
        const int last = (l == 4);
        combine<<<(nodeElems + NT - 1) / NT, NT, 0, stream>>>(h, agg, dis, b[l], nxt,
                                                              /*relu=*/!last, /*write_f32=*/last);
        __bf16* t = cur; cur = nxt; nxt = t;
    }

    // --- global mean pool + classifier (layer-5 fp32 result lives in agg) ---
    fill_f32<<<(N_GRAPHS * FEAT + NT - 1) / NT, NT, 0, stream>>>(pool, 0.0f, N_GRAPHS * FEAT);
    fill_f32<<<1, NT, 0, stream>>>(cnt, 0.0f, N_GRAPHS);
    pool_nodes<<<(N_NODES * 32 + NT - 1) / NT, NT, 0, stream>>>(agg, batch, pool, cnt);
    final_linear<<<(N_GRAPHS * N_CLASSES + NT - 1) / NT, NT, 0, stream>>>(pool, cnt, Wlin, blin, out);
}
